// NonLocalBlock2D_1760936591752
// MI455X (gfx1250) — matrix operand order repaired and checked by reading, compiled, not run on hardware
//
#include <hip/hip_runtime.h>
#include <math.h>

typedef unsigned short u16;
typedef unsigned int u32;
typedef __attribute__((ext_vector_type(16))) __bf16 v16bf;
typedef __attribute__((ext_vector_type(8)))  float  v8f;

union Frag { v16bf v; u32 u[8]; };

__device__ inline u16 f2bf(float x) {
    union { float f; u32 u; } c; c.f = x;
    u32 u = c.u;
    u32 r = u + 0x7FFFu + ((u >> 16) & 1u);   // round-to-nearest-even
    return (u16)(r >> 16);
}
__device__ inline float bf2f(u16 h) {
    union { u32 u; float f; } c; c.u = ((u32)h) << 16; return c.f;
}

__device__ inline v8f zero8() {
    v8f z = {0.f,0.f,0.f,0.f,0.f,0.f,0.f,0.f}; return z;
}

__device__ inline v8f wmma_bf16(v16bf a, v16bf b, v8f c) {
    // D = A(16x32 bf16) * B(32x16 bf16) + C(16x16 f32)
    return __builtin_amdgcn_wmma_f32_16x16x32_bf16(false, a, false, b, (short)0, c, false, false);
}

// A-matrix fragment (rows x K): lane l<16 -> row l, K in {0..7,16..23}; l>=16 -> row l-16, K in {8..15,24..31}
__device__ inline v16bf load_fragA(const __bf16* base, int stride, int row0, int k0, int lane) {
    const u32* p = (const u32*)(base + (size_t)(row0 + (lane & 15)) * stride + k0) + ((lane & 16) ? 4 : 0);
    Frag f;
    f.u[0]=p[0]; f.u[1]=p[1]; f.u[2]=p[2]; f.u[3]=p[3];
    f.u[4]=p[8]; f.u[5]=p[9]; f.u[6]=p[10]; f.u[7]=p[11];
    return f.v;
}

// B-matrix fragment (K x cols): lane l holds column l&15, K in {0..7,16..23} (l<16) or {8..15,24..31} (l>=16).
// "col0 + col" indexes rows of a row-major [N][K] array (i.e. B pre-transposed in memory).
__device__ inline v16bf load_fragB(const __bf16* base, int stride, int col0, int k0, int lane) {
    const u32* p = (const u32*)(base + (size_t)(col0 + (lane & 15)) * stride + k0) + ((lane & 16) ? 4 : 0);
    Frag f;
#pragma unroll
    for (int j = 0; j < 8; ++j) f.u[j] = p[j < 4 ? j : j + 4];
    return f.v;
}

// ---------------------------------------------------------------------------
// Problem constants
// ---------------------------------------------------------------------------
#define Bn 8
#define Cc 256
#define Ci 128
#define Hh 64
#define Ww 64
#define HW 4096
#define Np 1024
#define BN_EPS 1e-5f

// ---------------------------------------------------------------------------
// Kernel 0: convert weights to bf16; build theta_w^T [cin=256][c=128]
// ---------------------------------------------------------------------------
__global__ void __launch_bounds__(256)
convert_kernel(const float* phi_w, const float* g_w, const float* theta_w,
               u16* w_phi_bf, u16* w_g_bf, u16* theta_wT) {
    int i = blockIdx.x * 256 + threadIdx.x;
    if (i < Ci * Cc) {
        w_phi_bf[i] = f2bf(phi_w[i]);
        w_g_bf[i]   = f2bf(g_w[i]);
        int cin = i >> 7, c = i & 127;           // i = cin*128 + c
        theta_wT[i] = f2bf(theta_w[c * Cc + cin]);
    }
}

// ---------------------------------------------------------------------------
// Kernel 1: pooled projections phi_p, g_p = maxpool2(conv1x1(y, w)) + bias
// Workgroup: (batch, 8x8 spatial block). y tile staged to LDS as bf16.
// WMMA: A = y tile [pos=16 x c=32], B = w^T [c=32 x ci=16]; D rows are
// positions ordered quad-major so maxpool = per-lane max over acc VGPRs.
// ---------------------------------------------------------------------------
#define SY_STRIDE 130   // padded dword stride per position row (bank decorrelation)
__global__ void __launch_bounds__(256)
pool_proj_kernel(const float* __restrict__ y,
                 const u16* __restrict__ w_phi, const u16* __restrict__ w_g,
                 const float* __restrict__ phi_b, const float* __restrict__ g_b,
                 u16* __restrict__ phi_p, u16* __restrict__ g_p) {
    __shared__ u32 sy[64 * SY_STRIDE];           // 64 positions x 128 c-pairs (bf16x2)
    int b  = blockIdx.y;
    int sb = blockIdx.x;
    int h0 = (sb >> 3) * 8, w0 = (sb & 7) * 8;
    int tid = threadIdx.x, lane = tid & 31, wave = tid >> 5;
    const float* yb = y + (size_t)b * Cc * HW;

    // Stage y[b, :, 8x8 block] -> LDS bf16, positions in quad-major order.
    for (int i = tid; i < 8192; i += 256) {
        int j  = i & 63;                          // natural (dh,dw)
        int cp = i >> 6;                          // channel pair
        int dh = j >> 3, dw = j & 7;
        int quad   = (dh >> 1) * 4 + (dw >> 1);
        int within = (dh & 1) * 2 + (dw & 1);
        int pos = quad * 4 + within;
        size_t g0 = (size_t)(2 * cp) * HW + (size_t)(h0 + dh) * Ww + (w0 + dw);
        sy[pos * SY_STRIDE + cp] = (u32)f2bf(yb[g0]) | ((u32)f2bf(yb[g0 + HW]) << 16);
    }
    __syncthreads();

    // 16 tasks = {phi,g} x 8 ci-tiles; each wave takes 2.
#pragma unroll 1
    for (int task = wave; task < 16; task += 8) {
        int proj = task >> 3;
        int cit  = (task & 7) * 16;
        const __bf16* wsel = (const __bf16*)(proj ? w_g : w_phi);
        u16* outp  = proj ? g_p : phi_p;
        float bias = (proj ? g_b : phi_b)[cit + (lane & 15)];

        v16bf bfr[8];
#pragma unroll
        for (int k8 = 0; k8 < 8; ++k8)
            bfr[k8] = load_fragB(wsel, Cc, cit, k8 * 32, lane);

#pragma unroll 1
        for (int t = 0; t < 4; ++t) {            // 4 row tiles of 16 positions (4 quads each)
            v8f acc = zero8();
#pragma unroll
            for (int k8 = 0; k8 < 8; ++k8) {
                Frag a;
                const u32* sp = &sy[(t * 16 + (lane & 15)) * SY_STRIDE + k8 * 16 + ((lane & 16) ? 4 : 0)];
                a.u[0]=sp[0]; a.u[1]=sp[1]; a.u[2]=sp[2]; a.u[3]=sp[3];
                a.u[4]=sp[8]; a.u[5]=sp[9]; a.u[6]=sp[10]; a.u[7]=sp[11];
                acc = wmma_bf16(a.v, bfr[k8], acc);
            }
            // acc VGPRs 0..3 = rows {0..3|8..11} = quads 4t+{0|2}; 4..7 = quads 4t+{1|3}
            float m0 = fmaxf(fmaxf(acc[0], acc[1]), fmaxf(acc[2], acc[3])) + bias;
            float m1 = fmaxf(fmaxf(acc[4], acc[5]), fmaxf(acc[6], acc[7])) + bias;
            int q0 = 4 * t + ((lane & 16) ? 2 : 0);
            int q1 = q0 + 1;
            int ci = cit + (lane & 15);
            size_t base = ((size_t)b * Ci + ci) * Np;
            outp[base + ((h0 >> 1) + (q0 >> 2)) * 32 + ((w0 >> 1) + (q0 & 3))] = f2bf(m0);
            outp[base + ((h0 >> 1) + (q1 >> 2)) * 32 + ((w0 >> 1) + (q1 & 3))] = f2bf(m1);
        }
    }
}

// ---------------------------------------------------------------------------
// Kernel 2: S[b] = phi_p[b] (128xNp) @ g_p[b]^T (Npx128)  -> f32 [128x128]
// ---------------------------------------------------------------------------
__global__ void __launch_bounds__(256)
s_kernel(const u16* __restrict__ phi_p, const u16* __restrict__ g_p,
         float* __restrict__ S) {
    int b = blockIdx.x, tid = threadIdx.x, lane = tid & 31, wave = tid >> 5;
    const __bf16* A  = (const __bf16*)(phi_p + (size_t)b * Ci * Np);
    const __bf16* Bm = (const __bf16*)(g_p  + (size_t)b * Ci * Np);
    v8f acc[8];
#pragma unroll
    for (int n = 0; n < 8; ++n) acc[n] = zero8();

#pragma unroll 1
    for (int k8 = 0; k8 < Np / 32; ++k8) {
        v16bf a = load_fragA(A, Np, wave * 16, k8 * 32, lane);
#pragma unroll
        for (int n = 0; n < 8; ++n) {
            v16bf bf = load_fragB(Bm, Np, n * 16, k8 * 32, lane);
            acc[n] = wmma_bf16(a, bf, acc[n]);
        }
    }
    float* Sb = S + (size_t)b * Ci * Ci;
#pragma unroll
    for (int n = 0; n < 8; ++n)
#pragma unroll
        for (int r = 0; r < 8; ++r) {
            int row = wave * 16 + r + ((lane & 16) ? 8 : 0);
            int col = n * 16 + (lane & 15);
            Sb[row * Ci + col] = acc[n][r];
        }
}

// ---------------------------------------------------------------------------
// Kernel 3: T1t[b][co][c] = sum_c' S[b][c][c'] * W_w[co][c']   (bf16 out)
// ---------------------------------------------------------------------------
__global__ void __launch_bounds__(256)
t1_kernel(const float* __restrict__ S, const float* __restrict__ W_w,
          u16* __restrict__ T1t) {
    int idx = blockIdx.x * 256 + threadIdx.x;    // Bn*Ci*Cc = 262144
    int b = idx >> 15, rem = idx & 32767;
    int c = rem >> 8, co = rem & 255;
    const float* Sr = S + ((size_t)b * Ci + c) * Ci;
    const float* Wr = W_w + (size_t)co * Ci;
    float s = 0.f;
#pragma unroll 4
    for (int i = 0; i < Ci; ++i) s = fmaf(Sr[i], Wr[i], s);
    T1t[((size_t)b * Cc + co) * Ci + c] = f2bf(s);
}

// ---------------------------------------------------------------------------
// Kernel 4: bias[b][co] = bn_s*( (theta_b . T1t[b][co][:])/Np + W_b[co] ) + bn_beta
// ---------------------------------------------------------------------------
__global__ void __launch_bounds__(256)
bias_kernel(const u16* __restrict__ T1t, const float* __restrict__ theta_b,
            const float* __restrict__ W_b, const float* __restrict__ bn_g,
            const float* __restrict__ bn_b, float* __restrict__ biasv) {
    int b = blockIdx.x, co = threadIdx.x;
    const u16* T = T1t + ((size_t)b * Cc + co) * Ci;
    float s = 0.f;
#pragma unroll 4
    for (int c = 0; c < Ci; ++c) s = fmaf(theta_b[c], bf2f(T[c]), s);
    float bns = bn_g[co] / sqrtf(1.f + BN_EPS);
    biasv[b * Cc + co] = bns * (s * (1.f / (float)Np) + W_b[co]) + bn_b[co];
}

// ---------------------------------------------------------------------------
// Kernel 5: Kbt[b][co][cin] = bn_s[co]/Np * sum_c theta_wT[cin][c]*T1t[b][co][c]
// WMMA: A = theta_wT [cin x c], B = T1t rows (columns co), K = c = 128.
// ---------------------------------------------------------------------------
__global__ void __launch_bounds__(256)
kbt_kernel(const u16* __restrict__ theta_wT, const u16* __restrict__ T1t,
           const float* __restrict__ bn_g, u16* __restrict__ Kbt) {
    int b = blockIdx.x, tid = threadIdx.x, lane = tid & 31, wave = tid >> 5;
    const __bf16* A  = (const __bf16*)theta_wT;
    const __bf16* Bm = (const __bf16*)(T1t + (size_t)b * Cc * Ci);
    u16* Kb = Kbt + (size_t)b * Cc * Cc;

#pragma unroll 1
    for (int tile = wave; tile < 256; tile += 8) {   // 16 cin-tiles x 16 co-tiles
        int mt = tile >> 4, nt = tile & 15;
        v8f acc = zero8();
#pragma unroll
        for (int k8 = 0; k8 < 4; ++k8) {
            v16bf a  = load_fragA(A, Ci, mt * 16, k8 * 32, lane);
            v16bf bf = load_fragB(Bm, Ci, nt * 16, k8 * 32, lane);
            acc = wmma_bf16(a, bf, acc);
        }
        int co = nt * 16 + (lane & 15);
        float scale = bn_g[co] / sqrtf(1.f + BN_EPS) * (1.f / (float)Np);
#pragma unroll
        for (int r = 0; r < 8; ++r) {
            int cin = mt * 16 + r + ((lane & 16) ? 8 : 0);
            Kb[(size_t)co * Cc + cin] = f2bf(acc[r] * scale);   // stored transposed [co][cin]
        }
    }
}

// ---------------------------------------------------------------------------
// Kernel 6: out[b,co,n] = x[b,co,n] + sum_cin x[b,cin,n]*Kbt[b][co][cin] + bias[b][co]
// Workgroup: (batch, 128-position chunk); Kbt half staged in 64 KB LDS.
// WMMA: A = Kbt rows (co x cin) from LDS, B = x columns (gather+cvt), K = 256.
// ---------------------------------------------------------------------------
#define SK_STRIDE 132   // padded dword stride per co row
__global__ void __launch_bounds__(256)
final_kernel(const float* __restrict__ x, const u16* __restrict__ Kbt,
             const float* __restrict__ biasv, float* __restrict__ out) {
    __shared__ u32 sk[128 * SK_STRIDE];          // half of Kbt[b]: 128 co rows x 128 cin-pairs
    int b = blockIdx.y, chunk = blockIdx.x;
    int tid = threadIdx.x, lane = tid & 31, wave = tid >> 5;
    int n0 = chunk * 128 + wave * 16;
    const float* xb = x + (size_t)b * Cc * HW;

    // Build 8 B-fragments from x (gather along cin, coalesced along n).
    v16bf bfr[8];
    {
        int nn   = n0 + (lane & 15);
        int koff = (lane & 16) ? 8 : 0;
#pragma unroll
        for (int k8 = 0; k8 < 8; ++k8) {
            Frag f;
            int kb = k8 * 32 + koff;
#pragma unroll
            for (int j = 0; j < 8; ++j) {
                f.u[j] = (u32)f2bf(xb[(size_t)(kb + 2 * j + ((j & 4) ? 8 : 0)) * HW + nn]) |
                         ((u32)f2bf(xb[(size_t)(kb + 2 * j + ((j & 4) ? 8 : 0) + 1) * HW + nn]) << 16);
            }
            bfr[k8] = f.v;
        }
    }

    const u32* kg = (const u32*)(Kbt + (size_t)b * Cc * Cc);
#pragma unroll 1
    for (int half = 0; half < 2; ++half) {
        __syncthreads();
        for (int i = tid; i < 16384; i += 256) {
            int row = i >> 7, d = i & 127;
            sk[row * SK_STRIDE + d] = kg[half * 16384 + i];
        }
        __syncthreads();

#pragma unroll 1
        for (int mt = 0; mt < 8; ++mt) {
            int co0 = half * 128 + mt * 16;
            v8f acc = zero8();
#pragma unroll
            for (int k8 = 0; k8 < 8; ++k8) {
                Frag a;
                const u32* sp = &sk[(mt * 16 + (lane & 15)) * SK_STRIDE + k8 * 16 + ((lane & 16) ? 4 : 0)];
                a.u[0]=sp[0]; a.u[1]=sp[1]; a.u[2]=sp[2]; a.u[3]=sp[3];
                a.u[4]=sp[8]; a.u[5]=sp[9]; a.u[6]=sp[10]; a.u[7]=sp[11];
                acc = wmma_bf16(a.v, bfr[k8], acc);
            }
            int n = n0 + (lane & 15);
#pragma unroll
            for (int r = 0; r < 8; ++r) {
                int co = co0 + r + ((lane & 16) ? 8 : 0);
                size_t o = ((size_t)b * Cc + co) * HW + n;
                out[o] = x[o] + acc[r] + biasv[b * Cc + co];
            }
        }
    }
}

// ---------------------------------------------------------------------------
extern "C" void kernel_launch(void* const* d_in, const int* in_sizes, int n_in,
                              void* d_out, int out_size, void* d_ws, size_t ws_size,
                              hipStream_t stream) {
    const float* x       = (const float*)d_in[0];
    const float* y       = (const float*)d_in[1];
    const float* g_w     = (const float*)d_in[2];
    const float* g_b     = (const float*)d_in[3];
    const float* theta_w = (const float*)d_in[4];
    const float* theta_b = (const float*)d_in[5];
    const float* phi_w   = (const float*)d_in[6];
    const float* phi_b   = (const float*)d_in[7];
    const float* W_w     = (const float*)d_in[8];
    const float* W_b     = (const float*)d_in[9];
    const float* bn_g    = (const float*)d_in[10];
    const float* bn_b    = (const float*)d_in[11];

    char* ws = (char*)d_ws;
    u16*   w_phi_bf = (u16*)(ws);                               //  64 KB
    u16*   w_g_bf   = (u16*)(ws + 65536);                       //  64 KB
    u16*   theta_wT = (u16*)(ws + 131072);                      //  64 KB
    u16*   phi_p    = (u16*)(ws + 196608);                      //   2 MB
    u16*   g_p      = (u16*)(ws + 196608 + 2097152);            //   2 MB
    float* S        = (float*)(ws + 196608 + 4194304);          // 512 KB
    u16*   T1t      = (u16*)(ws + 196608 + 4194304 + 524288);   // 512 KB
    u16*   Kbt      = (u16*)(ws + 196608 + 4194304 + 1048576);  //   1 MB
    float* biasv    = (float*)(ws + 196608 + 4194304 + 2097152);//   8 KB

    convert_kernel<<<128, 256, 0, stream>>>(phi_w, g_w, theta_w, w_phi_bf, w_g_bf, theta_wT);
    pool_proj_kernel<<<dim3(64, Bn), 256, 0, stream>>>(y, w_phi_bf, w_g_bf, phi_b, g_b, phi_p, g_p);
    s_kernel<<<Bn, 256, 0, stream>>>(phi_p, g_p, S);
    t1_kernel<<<1024, 256, 0, stream>>>(S, W_w, T1t);
    bias_kernel<<<Bn, 256, 0, stream>>>(T1t, theta_b, W_b, bn_g, bn_b, biasv);
    kbt_kernel<<<Bn, 256, 0, stream>>>(theta_wT, T1t, bn_g, Kbt);
    final_kernel<<<dim3(32, Bn), 256, 0, stream>>>(x, Kbt, biasv, (float*)d_out);
}
